// SignedGCN_59270548685284
// MI455X (gfx1250) — compile-verified, matrix operand order checked by reading, not run on hardware
//
#include <hip/hip_runtime.h>
#include <hip/hip_bf16.h>

typedef __bf16 bf16_t;
typedef __attribute__((ext_vector_type(16))) __bf16 v16bf;
typedef __attribute__((ext_vector_type(8)))  float  v8f;

#define NNODES 100000
#define EPOS   1600000
#define ENEG   800000

static inline int cdiv(long long a, long long b) { return (int)((a + b - 1) / b); }

// ---------------- degree count (one thread per edge) ----------------
__global__ void degree_kernel(const int* __restrict__ dst, int E, float* __restrict__ deg) {
    int e = blockIdx.x * blockDim.x + threadIdx.x;
    if (e < E) atomicAdd(&deg[dst[e]], 1.0f);
}

// ---------------- deg -> 1/max(deg,1) ----------------
__global__ void recip_kernel(float* __restrict__ d, int n) {
    int i = blockIdx.x * blockDim.x + threadIdx.x;
    if (i < n) d[i] = 1.0f / fmaxf(d[i], 1.0f);
}

// ---------------- scatter-add: one wave per edge, 128 channels read,
// channels [0,64) -> agg col offA+c, channels [64,128) -> agg col offB+(c-64) ----------------
__global__ void scatter_add_kernel(const float* __restrict__ feat, int fstride,
                                   const int* __restrict__ src, const int* __restrict__ dst,
                                   int E, float* __restrict__ agg, int astride,
                                   int offA, int offB) {
    long long idx = (long long)blockIdx.x * blockDim.x + threadIdx.x;
    int e    = (int)(idx >> 5);
    int lane = (int)(idx & 31);
    if (e >= E) return;
    int s = src[e], d = dst[e];
    const float* fp = feat + (size_t)s * fstride;
    float*       op = agg  + (size_t)d * astride;
    atomicAdd(op + offA + lane,      fp[lane]);
    atomicAdd(op + offA + lane + 32, fp[lane + 32]);
    atomicAdd(op + offB + lane,      fp[lane + 64]);
    atomicAdd(op + offB + lane + 32, fp[lane + 96]);
}

// ---------------- pre-pack weights into WMMA B-fragment order, bf16 hi/lo split.
// Fragment index = (chunk*4 + t)*32 + lane; 16 bf16 elements per fragment-lane.
// Lane (half,m) of n-tile t holds col n = t*16+m, k = half*8+i (i<8), 16+half*8+(i-8) (i>=8). ----------------
__global__ void prep_w_kernel(const float* __restrict__ W, int nchunks,
                              bf16_t* __restrict__ Whi, bf16_t* __restrict__ Wlo) {
    int idx = blockIdx.x * blockDim.x + threadIdx.x;
    int total = nchunks * 4 * 32;
    if (idx >= total) return;
    int lane  = idx & 31;
    int t     = (idx >> 5) & 3;
    int chunk = idx >> 7;
    int m = lane & 15, half = lane >> 4;
    int n = t * 16 + m;
    bf16_t* ph = Whi + (size_t)idx * 16;
    bf16_t* pl = Wlo + (size_t)idx * 16;
#pragma unroll
    for (int i = 0; i < 16; ++i) {
        int k = (i < 8) ? (half * 8 + i) : (16 + half * 8 + (i - 8));
        float w = W[(size_t)(chunk * 32 + k) * 64 + n];
        bf16_t h = (bf16_t)w;
        ph[i] = h;
        pl[i] = (bf16_t)(w - (float)h);
    }
}

// ---------------- fused segmented GEMM with folded mean-normalization:
// out[:, off:off+64] = concat(r0.*A0, r1.*A1, r2.*A2) @ W + b
// B fragments pre-packed (Whi/Wlo). bf16 hi/lo split: 3 WMMAs per (chunk, n-tile). ----------------
__global__ void __launch_bounds__(256)
gemm_wmma_kernel(const float* __restrict__ a0, int s0, int c0, const float* __restrict__ r0,
                 const float* __restrict__ a1, int s1, int c1, const float* __restrict__ r1,
                 const float* __restrict__ a2, int s2, int c2, const float* __restrict__ r2,
                 const bf16_t* __restrict__ Whi, const bf16_t* __restrict__ Wlo,
                 const float* __restrict__ bias,
                 float* __restrict__ out, int out_stride, int out_coloff, int nrows) {
    int wave = (int)(((long long)blockIdx.x * blockDim.x + threadIdx.x) >> 5);
    int lane = threadIdx.x & 31;
    int rowBase = wave * 16;
    if (rowBase >= nrows) return;           // wave-uniform: EXEC stays all-ones for WMMA
    int m    = lane & 15;
    int half = lane >> 4;
    int K = c0 + c1 + c2;

    const v16bf* Bh = (const v16bf*)Whi;
    const v16bf* Bl = (const v16bf*)Wlo;

    v8f acc[4];
#pragma unroll
    for (int t = 0; t < 4; ++t) acc[t] = (v8f){0.f,0.f,0.f,0.f,0.f,0.f,0.f,0.f};

    int chunk = 0;
    for (int kk = 0; kk < K; kk += 32, ++chunk) {
        // locate segment (segments are multiples of 64, so a 32-chunk never straddles)
        const float* ap; const float* rs; int astr; int lk;
        if (kk < c0)            { ap = a0; rs = r0; astr = s0; lk = kk; }
        else if (kk < c0 + c1)  { ap = a1; rs = r1; astr = s1; lk = kk - c0; }
        else                    { ap = a2; rs = r2; astr = s2; lk = kk - c0 - c1; }

        int row = rowBase + m;
        float sc = rs ? rs[row] : 1.0f;

        // A fragment: lane (half,m) holds row M=m, k = half*8+i (i<8), 16+half*8+(i-8)
        const float* arow = ap + (size_t)row * astr + lk;
        float af[16];
        {
            const float4* p = (const float4*)(arow + half * 8);
            float4 q0 = p[0], q1 = p[1];
            af[0]=q0.x; af[1]=q0.y; af[2]=q0.z; af[3]=q0.w;
            af[4]=q1.x; af[5]=q1.y; af[6]=q1.z; af[7]=q1.w;
        }
        {
            const float4* p = (const float4*)(arow + 16 + half * 8);
            float4 q0 = p[0], q1 = p[1];
            af[8]=q0.x;  af[9]=q0.y;  af[10]=q0.z; af[11]=q0.w;
            af[12]=q1.x; af[13]=q1.y; af[14]=q1.z; af[15]=q1.w;
        }
        v16bf a_hi, a_lo;
#pragma unroll
        for (int i = 0; i < 16; ++i) {
            float v = af[i] * sc;
            bf16_t h = (bf16_t)v;
            a_hi[i] = h;
            a_lo[i] = (bf16_t)(v - (float)h);
        }

#pragma unroll
        for (int t = 0; t < 4; ++t) {
            int fidx = (chunk * 4 + t) * 32 + lane;
            v16bf b_hi = Bh[fidx];
            v16bf b_lo = Bl[fidx];
            acc[t] = __builtin_amdgcn_wmma_f32_16x16x32_bf16(false, a_hi, false, b_hi,
                                                             (short)0, acc[t], false, false);
            acc[t] = __builtin_amdgcn_wmma_f32_16x16x32_bf16(false, a_lo, false, b_hi,
                                                             (short)0, acc[t], false, false);
            acc[t] = __builtin_amdgcn_wmma_f32_16x16x32_bf16(false, a_hi, false, b_lo,
                                                             (short)0, acc[t], false, false);
        }
    }

    // Epilogue: D layout — lane(half,m): VGPR r holds (row = r + 8*half, col = m)
#pragma unroll
    for (int t = 0; t < 4; ++t) {
        int n = t * 16 + m;
        float bv = bias[n];
#pragma unroll
        for (int r = 0; r < 8; ++r) {
            int row = rowBase + r + 8 * half;
            out[(size_t)row * out_stride + out_coloff + n] = acc[t][r] + bv;
        }
    }
}

extern "C" void kernel_launch(void* const* d_in, const int* in_sizes, int n_in,
                              void* d_out, int out_size, void* d_ws, size_t ws_size,
                              hipStream_t stream) {
    const float* x   = (const float*)d_in[0];
    const int* pos   = (const int*)d_in[1];
    const int* neg   = (const int*)d_in[2];
    const float* w1p = (const float*)d_in[3],  *b1p = (const float*)d_in[4];
    const float* w1n = (const float*)d_in[5],  *b1n = (const float*)d_in[6];
    const float* w2p = (const float*)d_in[7],  *b2p = (const float*)d_in[8];
    const float* w2n = (const float*)d_in[9],  *b2n = (const float*)d_in[10];
    const float* w3p = (const float*)d_in[11], *b3p = (const float*)d_in[12];
    const float* w3n = (const float*)d_in[13], *b3n = (const float*)d_in[14];
    float* out = (float*)d_out;

    const int N = NNODES, EP = EPOS, EN = ENEG;

    // ---- workspace layout: bf16 weight-fragment region first (32B aligned), floats after ----
    const int FR1 = 8 * 4 * 32 * 16;   // K=256: 16384 bf16 per (matrix, half)
    const int FR2 = 6 * 4 * 32 * 16;   // K=192: 12288 bf16 per (matrix, half)
    bf16_t* frag = (bf16_t*)d_ws;
    bf16_t* w1p_h = frag;               bf16_t* w1p_l = w1p_h + FR1;
    bf16_t* w1n_h = w1p_l + FR1;        bf16_t* w1n_l = w1n_h + FR1;
    bf16_t* w2p_h = w1n_l + FR1;        bf16_t* w2p_l = w2p_h + FR2;
    bf16_t* w2n_h = w2p_l + FR2;        bf16_t* w2n_l = w2n_h + FR2;
    bf16_t* w3p_h = w2n_l + FR2;        bf16_t* w3p_l = w3p_h + FR2;
    bf16_t* w3n_h = w3p_l + FR2;        bf16_t* w3n_l = w3n_h + FR2;

    float* ws     = (float*)(w3n_l + FR2);   // 327680 bytes in -> still 256B aligned
    float* rdeg_p = ws;
    float* rdeg_n = ws + N;
    float* agg    = ws + 2 * (size_t)N;
    float* z1     = agg + (size_t)N * 256;
    float* z2     = z1  + (size_t)N * 128;

    const int TB = 256;
    const int gemm_blocks = cdiv(N / 16, 8);          // 8 waves (16-row tiles) per block
    const int sc_blocks_p = cdiv((long long)EP * 32, TB);
    const int sc_blocks_n = cdiv((long long)EN * 32, TB);

    // ---- weight fragment prep (tiny; L2-resident afterwards) ----
    prep_w_kernel<<<cdiv(8*4*32, TB), TB, 0, stream>>>(w1p, 8, w1p_h, w1p_l);
    prep_w_kernel<<<cdiv(8*4*32, TB), TB, 0, stream>>>(w1n, 8, w1n_h, w1n_l);
    prep_w_kernel<<<cdiv(6*4*32, TB), TB, 0, stream>>>(w2p, 6, w2p_h, w2p_l);
    prep_w_kernel<<<cdiv(6*4*32, TB), TB, 0, stream>>>(w2n, 6, w2n_h, w2n_l);
    prep_w_kernel<<<cdiv(6*4*32, TB), TB, 0, stream>>>(w3p, 6, w3p_h, w3p_l);
    prep_w_kernel<<<cdiv(6*4*32, TB), TB, 0, stream>>>(w3n, 6, w3n_h, w3n_l);

    // ---- degrees (once, reused by all layers) ----
    hipMemsetAsync(rdeg_p, 0, 2 * (size_t)N * sizeof(float), stream);
    degree_kernel<<<cdiv(EP, TB), TB, 0, stream>>>(pos + EP, EP, rdeg_p);
    degree_kernel<<<cdiv(EN, TB), TB, 0, stream>>>(neg + EN, EN, rdeg_n);
    recip_kernel<<<cdiv(2 * N, TB), TB, 0, stream>>>(rdeg_p, 2 * N);

    // ======== Layer 1: z1 = [lin([mean_p(x), x])@w1p+b1p | lin([mean_n(x), x])@w1n+b1n] ========
    hipMemsetAsync(agg, 0, (size_t)N * 256 * sizeof(float), stream);
    scatter_add_kernel<<<sc_blocks_p, TB, 0, stream>>>(x, 128, pos, pos + EP, EP, agg, 256,   0,  64);
    scatter_add_kernel<<<sc_blocks_n, TB, 0, stream>>>(x, 128, neg, neg + EN, EN, agg, 256, 128, 192);
    gemm_wmma_kernel<<<gemm_blocks, TB, 0, stream>>>(
        agg,       256, 128, rdeg_p,   x, 128, 128, nullptr,   nullptr, 0, 0, nullptr,
        w1p_h, w1p_l, b1p, z1, 128,  0, N);
    gemm_wmma_kernel<<<gemm_blocks, TB, 0, stream>>>(
        agg + 128, 256, 128, rdeg_n,   x, 128, 128, nullptr,   nullptr, 0, 0, nullptr,
        w1n_h, w1n_l, b1n, z1, 128, 64, N);

    // ======== Layer 2: z2 ========
    // agg cols: [0:64)=sum_pp, [64:128)=sum_nn, [128:192)=sum_np, [192:256)=sum_pn
    hipMemsetAsync(agg, 0, (size_t)N * 256 * sizeof(float), stream);
    scatter_add_kernel<<<sc_blocks_p, TB, 0, stream>>>(z1, 128, pos, pos + EP, EP, agg, 256,   0, 128); // z_p->pp, z_n->np
    scatter_add_kernel<<<sc_blocks_n, TB, 0, stream>>>(z1, 128, neg, neg + EN, EN, agg, 256, 192,  64); // z_p->pn, z_n->nn
    gemm_wmma_kernel<<<gemm_blocks, TB, 0, stream>>>(
        agg,       256, 64, rdeg_p,   agg + 64,  256, 64, rdeg_n,   z1,      128, 64, nullptr,
        w2p_h, w2p_l, b2p, z2, 128,  0, N);
    gemm_wmma_kernel<<<gemm_blocks, TB, 0, stream>>>(
        agg + 128, 256, 64, rdeg_p,   agg + 192, 256, 64, rdeg_n,   z1 + 64, 128, 64, nullptr,
        w2n_h, w2n_l, b2n, z2, 128, 64, N);

    // ======== Layer 3: out ========
    hipMemsetAsync(agg, 0, (size_t)N * 256 * sizeof(float), stream);
    scatter_add_kernel<<<sc_blocks_p, TB, 0, stream>>>(z2, 128, pos, pos + EP, EP, agg, 256,   0, 128);
    scatter_add_kernel<<<sc_blocks_n, TB, 0, stream>>>(z2, 128, neg, neg + EN, EN, agg, 256, 192,  64);
    gemm_wmma_kernel<<<gemm_blocks, TB, 0, stream>>>(
        agg,       256, 64, rdeg_p,   agg + 64,  256, 64, rdeg_n,   z2,      128, 64, nullptr,
        w3p_h, w3p_l, b3p, out, 128,  0, N);
    gemm_wmma_kernel<<<gemm_blocks, TB, 0, stream>>>(
        agg + 128, 256, 64, rdeg_p,   agg + 192, 256, 64, rdeg_n,   z2 + 64, 128, 64, nullptr,
        w3n_h, w3n_l, b3n, out, 128, 64, N);
}
